// AttentionAgent_4234837754412
// MI455X (gfx1250) — matrix operand order repaired and checked by reading, not compiled, not run on hardware
//
#include <hip/hip_runtime.h>
#include <hip/hip_bf16.h>
#include <math.h>

typedef _Float16 half_t;
typedef __attribute__((ext_vector_type(16))) _Float16 v16h;
typedef __attribute__((ext_vector_type(8)))  _Float16 v8h;
typedef __attribute__((ext_vector_type(8)))  float    v8f;

#define N_TRAJ   16
#define HEADS    8
#define HIDDEN   512
#define ATT_D    64
#define STATE_DIM 128
#define ACTION_DIM 18
#define TRAJ_DIM 9
#define BATCH    4096

// ---------------- WMMA GEMM: C(f16) = act(A(f16)[M,K] @ B) ----------------
// B is PRE-PACKED: Bp[((kt*Npack + n) << 5) + kk]  (kt = k/32, kk = k%32).
// block = 256 threads = 8 waves; tile 128x128, K-step 32; double-buffered LDS;
// batched via gridDim.z.
#define TM 128
#define TN 128
#define LDSD 40            // padded LDS row stride (f16) -> conflict-free b128 frags
#define LDS_BUF (2 * TM * LDSD)

static __device__ inline v16h cat16(v8h lo, v8h hi) {
  return __builtin_shufflevector(lo, hi, 0,1,2,3,4,5,6,7,8,9,10,11,12,13,14,15);
}

__global__ __launch_bounds__(256)
void wmma_gemm_f16(const half_t* __restrict__ A, const half_t* __restrict__ Bp,
                   const float* __restrict__ bias, half_t* __restrict__ C,
                   int K, int lda, int nPack, int ldc,
                   long sA, long sB, long sBias, long sC, int act)
{
  __shared__ half_t lds[2 * LDS_BUF];   // [buffer][A 128x40 | B 128x40]

  const int z = blockIdx.z;
  const half_t* Ab = A + (long)z * sA;
  const half_t* Bb = Bp + (long)z * sB;
  const float*  Gb = bias ? (bias + (long)z * sBias) : (const float*)0;
  half_t* Cb = C + (long)z * sC;

  const int n0 = blockIdx.x * TN;
  const int m0 = blockIdx.y * TM;
  const int tid = threadIdx.x;
  const int w   = tid >> 5;
  const int l   = tid & 31;
  const int lm  = l & 15;
  const int lh  = l >> 4;
  const int mb  = (w >> 1) * 32;     // wave row base
  const int nb  = (w & 1) * 64;      // wave col base

  // per-thread staging coords: 512 chunks of 8 f16 cover a 128x32 tile
  const int ch0 = tid,       ch1 = tid + 256;
  const int r0 = ch0 >> 2,   c0 = (ch0 & 3) * 8;
  const int r1 = ch1 >> 2,   c1 = (ch1 & 3) * 8;

  const int nkt = K >> 5;
  v8f acc[2][4] = {};

  auto aSrc = [&](int kt, int r, int c8) -> const v8h* {
    return (const v8h*)(Ab + (long)(m0 + r) * lda + (kt << 5) + c8);
  };
  auto bSrc = [&](int kt, int r, int c8) -> const v8h* {
    return (const v8h*)(Bb + (((long)kt * nPack + n0 + r) << 5) + c8);
  };

  // ---- preload tile 0
  {
    half_t* la = lds;
    half_t* lb = lds + TM * LDSD;
    *(v8h*)(la + r0 * LDSD + c0) = *aSrc(0, r0, c0);
    *(v8h*)(la + r1 * LDSD + c1) = *aSrc(0, r1, c1);
    *(v8h*)(lb + r0 * LDSD + c0) = *bSrc(0, r0, c0);
    *(v8h*)(lb + r1 * LDSD + c1) = *bSrc(0, r1, c1);
  }
  __syncthreads();

  for (int kt = 0; kt < nkt; ++kt) {
    half_t* la = lds + (kt & 1) * LDS_BUF;
    half_t* lb = la + TM * LDSD;

    // ---- issue next tile's global loads early (latency hides under WMMA)
    v8h a0{}, a1{}, b0{}, b1{};
    const bool more = (kt + 1) < nkt;
    if (more) {
      a0 = *aSrc(kt + 1, r0, c0);
      a1 = *aSrc(kt + 1, r1, c1);
      b0 = *bSrc(kt + 1, r0, c0);
      b1 = *bSrc(kt + 1, r1, c1);
      if (kt + 2 < nkt) {                 // gfx1250 global_prefetch path
        __builtin_prefetch((const void*)aSrc(kt + 2, r0, c0), 0, 0);
        __builtin_prefetch((const void*)bSrc(kt + 2, r0, c0), 0, 0);
      }
    }

    // ---- fragments (ISA 16-bit A 16x32 / B 32x16 layouts; two b128 reads each)
    v16h af[2], bf[4];
#pragma unroll
    for (int ti = 0; ti < 2; ti++) {
      const half_t* p = la + (mb + 16 * ti + lm) * LDSD + lh * 8;
      af[ti] = cat16(*(const v8h*)p, *(const v8h*)(p + 16));
    }
#pragma unroll
    for (int tj = 0; tj < 4; tj++) {
      const half_t* p = lb + (nb + 16 * tj + lm) * LDSD + lh * 8;
      bf[tj] = cat16(*(const v8h*)p, *(const v8h*)(p + 16));
    }
#pragma unroll
    for (int ti = 0; ti < 2; ti++)
#pragma unroll
      for (int tj = 0; tj < 4; tj++)
        acc[ti][tj] = __builtin_amdgcn_wmma_f32_16x16x32_f16(
            false, af[ti], false, bf[tj], (short)0, acc[ti][tj], false, false);

    // ---- stage next tile into the other LDS buffer; single barrier per step
    if (more) {
      half_t* na  = lds + ((kt + 1) & 1) * LDS_BUF;
      half_t* nb2 = na + TM * LDSD;
      *(v8h*)(na  + r0 * LDSD + c0) = a0;
      *(v8h*)(na  + r1 * LDSD + c1) = a1;
      *(v8h*)(nb2 + r0 * LDSD + c0) = b0;
      *(v8h*)(nb2 + r1 * LDSD + c1) = b1;
      __syncthreads();
    }
  }

  // ---- epilogue: bias + activation, store f16
#pragma unroll
  for (int ti = 0; ti < 2; ti++) {
#pragma unroll
    for (int tj = 0; tj < 4; tj++) {
      int n = n0 + nb + 16 * tj + lm;
      float bv = Gb ? Gb[n] : 0.0f;
#pragma unroll
      for (int r = 0; r < 8; r++) {
        int m = m0 + mb + 16 * ti + lh * 8 + r;
        float v = acc[ti][tj][r] + bv;
        if (act == 1)      v = v > 0.f ? v : 0.f;
        else if (act == 2) v = v > 0.f ? v : 0.01f * v;
        Cb[(long)m * ldc + n] = (half_t)v;
      }
    }
  }
}

// ---------------- packing / conversion kernels ----------------
__global__ void k_f32_to_f16(half_t* __restrict__ dst, const float* __restrict__ src, int n) {
  int i = blockIdx.x * 256 + threadIdx.x;
  if (i < n) dst[i] = (half_t)src[i];
}

// pack weights [batch][r..K rows][N] f32 -> Bp[batch][kt][n][kk] f16, zero-pad rows >= r
__global__ void k_pack_B(half_t* __restrict__ dst, const float* __restrict__ src,
                         int K, int N, int r, long total) {
  long i = (long)blockIdx.x * 256 + threadIdx.x;
  if (i >= total) return;
  int  kk = (int)(i & 31);
  long t  = i >> 5;
  int  n  = (int)(t % N);
  long t2 = t / N;
  int  nkt = K >> 5;
  int  kt = (int)(t2 % nkt);
  int  b  = (int)(t2 / nkt);
  int  k  = (kt << 5) + kk;
  dst[i] = (k < r) ? (half_t)src[((long)b * r + k) * N + n] : (half_t)0.0f;
}

// (HEADS,HIDDEN,ATT_D) f32 -> packed Bp for [HIDDEN][HEADS*ATT_D] (n = head*64+d)
__global__ void k_pack_heads(half_t* __restrict__ dst, const float* __restrict__ src) {
  int i = blockIdx.x * 256 + threadIdx.x;
  if (i >= HIDDEN * HIDDEN) return;
  int kk = i & 31;
  int n  = (i >> 5) & 511;
  int kt = i >> 14;
  int c  = (kt << 5) + kk;
  int head = n >> 6, d = n & 63;
  dst[i] = (half_t)src[((long)head * HIDDEN + c) * ATT_D + d];
}

// action head (K=512 -> 18), tanh+clip, cumsum mean, padded traj (t dim 9 -> 32)
__global__ void k_action_traj(const half_t* __restrict__ h3, const float* __restrict__ W4,
                              const float* __restrict__ b4, const float* __restrict__ noise,
                              half_t* __restrict__ trajp) {
  int b = blockIdx.x * 256 + threadIdx.x;
  if (b >= BATCH) return;
  float acc[18];
#pragma unroll
  for (int a = 0; a < 18; a++) acc[a] = b4[a];
  const half_t* hr = h3 + (long)b * HIDDEN;
  for (int c = 0; c < HIDDEN; c++) {
    float hv = (float)hr[c];
    const float* wr = W4 + c * 18;
#pragma unroll
    for (int a = 0; a < 18; a++) acc[a] += hv * wr[a];
  }
  float act[18];
#pragma unroll
  for (int a = 0; a < 18; a++) {
    float v = tanhf(acc[a]);
    float lo = (a < 9) ? -1.0f : 0.0f;
    act[a] = fminf(1.0f, fmaxf(lo, v));
  }
  float mean[9];
#pragma unroll
  for (int c2 = 0; c2 < 3; c2++) {
    mean[c2]     = act[c2];
    mean[3 + c2] = mean[c2] + act[3 + c2];
    mean[6 + c2] = mean[3 + c2] + act[6 + c2];
  }
  for (int n = 0; n < N_TRAJ; n++) {
    half_t* dst = trajp + ((long)n * BATCH + b) * 32;
    const float* nz = noise + ((long)n * BATCH + b) * TRAJ_DIM;
#pragma unroll
    for (int t = 0; t < 9; t++) dst[t] = (half_t)(mean[t] + act[9 + t] * nz[t]);
#pragma unroll
    for (int t = 9; t < 32; t++) dst[t] = (half_t)0.0f;
  }
}

// attention with the reference's flat-reshape gather semantics.
__global__ __launch_bounds__(128)
void k_attention(const half_t* __restrict__ keys, const half_t* __restrict__ vals,
                 const half_t* __restrict__ sels, half_t* __restrict__ cin) {
  int b = blockIdx.x;
  int t = threadIdx.x;
  int k = t >> 4, n = t & 15;
  float sel[64];
  const half_t* sp = sels + ((long)n * BATCH + b) * HIDDEN + k * ATT_D;
  for (int d = 0; d < 64; d++) sel[d] = (float)sp[d];
  float logits[15];
  for (int m = 0; m < 15; m++) {
    float a = 0.f;
    for (int d = 0; d < 64; d++) {
      int s  = 15 * d + m;
      int e  = s >> 6, dp = s & 63;
      int bb = 15 * b + e;
      int bp = bb & 4095;
      int mp = bb >> 12;
      int j  = mp + (mp >= n ? 1 : 0);
      a += sel[d] * (float)keys[((long)j * BATCH + bp) * HIDDEN + k * ATT_D + dp];
    }
    logits[m] = a * 0.125f;   // 1/sqrt(64)
  }
  float mx = logits[0];
  for (int m = 1; m < 15; m++) mx = fmaxf(mx, logits[m]);
  float wgt[15], sum = 0.f;
  for (int m = 0; m < 15; m++) { wgt[m] = __expf(logits[m] - mx); sum += wgt[m]; }
  float inv = 1.0f / sum;
  half_t* op = cin + ((long)n * BATCH + b) * (2 * HIDDEN) + HIDDEN + k * ATT_D;
  for (int d = 0; d < 64; d++) {
    float o = 0.f;
    for (int m = 0; m < 15; m++) {
      int s  = 15 * d + m;
      int e  = s >> 6, dp = s & 63;
      int bb = 15 * b + e;
      int bp = bb & 4095;
      int mp = bb >> 12;
      int j  = mp + (mp >= n ? 1 : 0);
      o += wgt[m] * inv * (float)vals[((long)j * BATCH + bp) * HIDDEN + k * ATT_D + dp];
    }
    op[d] = (half_t)o;
  }
}

// q head (N=1 per n) + scrambled q.reshape(B,16) into K-padded policy input (4096x32)
__global__ void k_q(const half_t* __restrict__ c1, const float* __restrict__ W2,
                    const float* __restrict__ b2, half_t* __restrict__ polp) {
  int g = blockIdx.x * 256 + threadIdx.x;   // g = n*4096 + b
  if (g >= N_TRAJ * BATCH) return;
  int n = g >> 12;
  float a = b2[n];
  const half_t* cp = c1 + (long)g * HIDDEN;
  const float*  wp = W2 + n * HIDDEN;
  for (int h = 0; h < HIDDEN; h++) a += (float)cp[h] * wp[h];
  int i = g >> 4, j = g & 15;               // flat remap (reference reshape)
  polp[i * 32 + j]      = (half_t)a;
  polp[i * 32 + 16 + j] = (half_t)0.0f;     // K padding
}

// final head: tanh(t2 @ traj_W3 + b3) * 1.0 -> f32 out (4096x9)
__global__ void k_final(const half_t* __restrict__ t2, const float* __restrict__ W3,
                        const float* __restrict__ b3, float* __restrict__ out) {
  int b = blockIdx.x * 256 + threadIdx.x;
  if (b >= BATCH) return;
  float acc[9];
#pragma unroll
  for (int t = 0; t < 9; t++) acc[t] = b3[t];
  const half_t* tp = t2 + (long)b * 256;
  for (int h = 0; h < 256; h++) {
    float hv = (float)tp[h];
    const float* wr = W3 + h * 9;
#pragma unroll
    for (int t = 0; t < 9; t++) acc[t] += hv * wr[t];
  }
#pragma unroll
  for (int t = 0; t < 9; t++) out[b * 9 + t] = tanhf(acc[t]);
}

// ---------------- launcher ----------------
extern "C" void kernel_launch(void* const* d_in, const int* in_sizes, int n_in,
                              void* d_out, int out_size, void* d_ws, size_t ws_size,
                              hipStream_t stream) {
  const float* state     = (const float*)d_in[0];
  const float* noise     = (const float*)d_in[1];
  const float* net_W1    = (const float*)d_in[2];
  const float* net_b1    = (const float*)d_in[3];
  const float* net_W2    = (const float*)d_in[4];
  const float* net_b2    = (const float*)d_in[5];
  const float* net_W3    = (const float*)d_in[6];
  const float* net_b3    = (const float*)d_in[7];
  const float* net_W4    = (const float*)d_in[8];
  const float* net_b4    = (const float*)d_in[9];
  const float* enc_W     = (const float*)d_in[10];
  const float* enc_b     = (const float*)d_in[11];
  const float* senc_W    = (const float*)d_in[12];
  const float* senc_b    = (const float*)d_in[13];
  const float* key_W     = (const float*)d_in[14];
  const float* sel_W     = (const float*)d_in[15];
  const float* val_W     = (const float*)d_in[16];
  const float* val_b     = (const float*)d_in[17];
  const float* critic_W1 = (const float*)d_in[18];
  const float* critic_b1 = (const float*)d_in[19];
  const float* critic_W2 = (const float*)d_in[20];
  const float* critic_b2 = (const float*)d_in[21];
  const float* traj_W1   = (const float*)d_in[22];
  const float* traj_b1   = (const float*)d_in[23];
  const float* traj_W2   = (const float*)d_in[24];
  const float* traj_b2   = (const float*)d_in[25];
  const float* traj_W3   = (const float*)d_in[26];
  const float* traj_b3   = (const float*)d_in[27];
  float* out = (float*)d_out;

  // ---- arena in d_ws (f16 buffers), 256B aligned
  char* ws = (char*)d_ws;
  size_t off = 0;
  auto alloc_h = [&](size_t elems) -> half_t* {
    off = (off + 255) & ~(size_t)255;
    half_t* p = (half_t*)(ws + off);
    off += elems * sizeof(half_t);
    return p;
  };
  half_t* state_h = alloc_h((size_t)BATCH * STATE_DIM);
  half_t* W1p  = alloc_h((size_t)STATE_DIM * HIDDEN);
  half_t* W2p  = alloc_h((size_t)HIDDEN * HIDDEN);
  half_t* W3p  = alloc_h((size_t)HIDDEN * HIDDEN);
  half_t* h1   = alloc_h((size_t)BATCH * HIDDEN);
  half_t* h2   = alloc_h((size_t)BATCH * HIDDEN);
  half_t* h3   = alloc_h((size_t)BATCH * HIDDEN);
  half_t* trajp  = alloc_h((size_t)N_TRAJ * BATCH * 32);
  half_t* encWp  = alloc_h((size_t)N_TRAJ * 32 * HIDDEN);
  half_t* sencWp = alloc_h((size_t)N_TRAJ * 32 * HIDDEN);
  half_t* Bkey = alloc_h((size_t)HIDDEN * HIDDEN);
  half_t* Bsel = alloc_h((size_t)HIDDEN * HIDDEN);
  half_t* Bval = alloc_h((size_t)HIDDEN * HIDDEN);
  half_t* cW1p = alloc_h((size_t)N_TRAJ * 2 * HIDDEN * HIDDEN);
  half_t* tW1p = alloc_h((size_t)32 * 256);
  half_t* tW2p = alloc_h((size_t)256 * 256);
  half_t* ta    = alloc_h((size_t)N_TRAJ * BATCH * HIDDEN);   // ta_enc, reused for c1
  half_t* keysB = alloc_h((size_t)N_TRAJ * BATCH * HIDDEN);
  half_t* valsB = alloc_h((size_t)N_TRAJ * BATCH * HIDDEN);
  half_t* selsB = alloc_h((size_t)N_TRAJ * BATCH * HIDDEN);
  half_t* cin   = alloc_h((size_t)N_TRAJ * BATCH * 2 * HIDDEN);
  half_t* polp  = alloc_h((size_t)BATCH * 32);
  half_t* t1    = alloc_h((size_t)BATCH * 256);
  half_t* t2    = alloc_h((size_t)BATCH * 256);
  (void)ws_size; (void)n_in; (void)in_sizes; (void)out_size;

  auto packB = [&](half_t* d, const float* s, int batch, int K, int N, int r) {
    long total = (long)batch * K * N;
    k_pack_B<<<(int)((total + 255) / 256), 256, 0, stream>>>(d, s, K, N, r, total);
  };
  // ---- conversions / weight packing (one-time, WMMA-frag-friendly layout)
  k_f32_to_f16<<<(BATCH * STATE_DIM + 255) / 256, 256, 0, stream>>>(
      state_h, state, BATCH * STATE_DIM);
  packB(W1p,  net_W1,  1, STATE_DIM, HIDDEN, STATE_DIM);
  packB(W2p,  net_W2,  1, HIDDEN, HIDDEN, HIDDEN);
  packB(W3p,  net_W3,  1, HIDDEN, HIDDEN, HIDDEN);
  packB(cW1p, critic_W1, N_TRAJ, 2 * HIDDEN, HIDDEN, 2 * HIDDEN);
  packB(tW2p, traj_W2, 1, 256, 256, 256);
  packB(encWp,  enc_W,  N_TRAJ, 32, HIDDEN, TRAJ_DIM);
  packB(sencWp, senc_W, N_TRAJ, 32, HIDDEN, TRAJ_DIM);
  packB(tW1p, traj_W1, 1, 32, 256, 16);
  {
    int n3 = HIDDEN * HIDDEN;
    k_pack_heads<<<(n3 + 255) / 256, 256, 0, stream>>>(Bkey, key_W);
    k_pack_heads<<<(n3 + 255) / 256, 256, 0, stream>>>(Bsel, sel_W);
    k_pack_heads<<<(n3 + 255) / 256, 256, 0, stream>>>(Bval, val_W);
  }

  auto gemm = [&](const half_t* A, const half_t* B, const float* bias, half_t* C,
                  int M, int N, int K, int lda, int ldc,
                  long sA, long sB, long sBias, long sC, int act, int batch) {
    dim3 grid(N / TN, M / TM, batch);
    wmma_gemm_f16<<<grid, 256, 0, stream>>>(A, B, bias, C, K, lda, N, ldc,
                                            sA, sB, sBias, sC, act);
  };

  // ---- policy MLP (relu)
  gemm(state_h, W1p, net_b1, h1, BATCH, HIDDEN, STATE_DIM, STATE_DIM, HIDDEN,
       0, 0, 0, 0, 1, 1);
  gemm(h1, W2p, net_b2, h2, BATCH, HIDDEN, HIDDEN, HIDDEN, HIDDEN, 0, 0, 0, 0, 1, 1);
  gemm(h2, W3p, net_b3, h3, BATCH, HIDDEN, HIDDEN, HIDDEN, HIDDEN, 0, 0, 0, 0, 1, 1);

  // ---- action -> padded trajectories
  k_action_traj<<<(BATCH + 255) / 256, 256, 0, stream>>>(h3, net_W4, net_b4, noise, trajp);

  // ---- encoders (lrelu), batched over n (K=32 padded)
  gemm(trajp, encWp, enc_b, ta, BATCH, HIDDEN, 32, 32, HIDDEN,
       (long)BATCH * 32, (long)32 * HIDDEN, HIDDEN, (long)BATCH * HIDDEN, 2, N_TRAJ);
  gemm(trajp, sencWp, senc_b, cin, BATCH, HIDDEN, 32, 32, 2 * HIDDEN,
       (long)BATCH * 32, (long)32 * HIDDEN, HIDDEN, (long)BATCH * 2 * HIDDEN, 2, N_TRAJ);

  // ---- keys / vals (from ta_enc), sels (from s_enc slice of cin); shared B across n
  gemm(ta, Bkey, nullptr, keysB, BATCH, HIDDEN, HIDDEN, HIDDEN, HIDDEN,
       (long)BATCH * HIDDEN, 0, 0, (long)BATCH * HIDDEN, 0, N_TRAJ);
  gemm(ta, Bval, val_b, valsB, BATCH, HIDDEN, HIDDEN, HIDDEN, HIDDEN,
       (long)BATCH * HIDDEN, 0, 0, (long)BATCH * HIDDEN, 2, N_TRAJ);
  gemm(cin, Bsel, nullptr, selsB, BATCH, HIDDEN, HIDDEN, 2 * HIDDEN, HIDDEN,
       (long)BATCH * 2 * HIDDEN, 0, 0, (long)BATCH * HIDDEN, 0, N_TRAJ);

  // ---- attention (exact flat-reshape semantics) -> critic_in[:, :, 512:1024]
  k_attention<<<BATCH, 128, 0, stream>>>(keysB, valsB, selsB, cin);

  // ---- critic layer 1 (lrelu), K=1024, batched weights; c1 reuses ta buffer
  gemm(cin, cW1p, critic_b1, ta, BATCH, HIDDEN, 2 * HIDDEN, 2 * HIDDEN, HIDDEN,
       (long)BATCH * 2 * HIDDEN, (long)2 * HIDDEN * HIDDEN, HIDDEN,
       (long)BATCH * HIDDEN, 2, N_TRAJ);

  // ---- q head + scrambled reshape into padded policy input
  k_q<<<(N_TRAJ * BATCH + 255) / 256, 256, 0, stream>>>(ta, critic_W2, critic_b2, polp);

  // ---- trajectory MLP
  gemm(polp, tW1p, traj_b1, t1, BATCH, 256, 32, 32, 256, 0, 0, 0, 0, 1, 1);
  gemm(t1, tW2p, traj_b2, t2, BATCH, 256, 256, 256, 256, 0, 0, 0, 0, 1, 1);
  k_final<<<(BATCH + 255) / 256, 256, 0, stream>>>(t2, traj_W3, traj_b3, out);
}
